// CausalSelfAttention_44753559224660
// MI455X (gfx1250) — compile-verified, matrix operand order checked by reading, not run on hardware
//
#include <hip/hip_runtime.h>
#include <hip/hip_bf16.h>

typedef __bf16 bf16_t;
typedef bf16_t v16bf __attribute__((ext_vector_type(16)));
typedef bf16_t v8bf  __attribute__((ext_vector_type(8)));
typedef float  v8f   __attribute__((ext_vector_type(8)));

#define D_MODEL  1024
#define N_HEADS  16
#define HEAD_DIM 64
#define BATCH    2
#define TQ       2048
#define TCACHE   2048
#define TK       4096
#define MROWS    (BATCH * TQ)   // 4096

// native f32 -> bf16 (RNE); lowers to the hardware convert on gfx1250
static __device__ __forceinline__ bf16_t f2bf(float f) { return (bf16_t)f; }

static __device__ __forceinline__ v8f vzero8() {
    v8f z;
#pragma unroll
    for (int i = 0; i < 8; ++i) z[i] = 0.0f;
    return z;
}

static __device__ __forceinline__ v16bf join8(v8bf lo, v8bf hi) {
    return __builtin_shufflevector(lo, hi, 0,1,2,3,4,5,6,7,8,9,10,11,12,13,14,15);
}

// DPP row-rotation (within 16-lane row) — pure VALU, co-executes with WMMA
template <int CTRL>
static __device__ __forceinline__ float fdpp(float x) {
    int i = __builtin_bit_cast(int, x);
    int r = __builtin_amdgcn_update_dpp(i, i, CTRL, 0xF, 0xF, false);
    return __builtin_bit_cast(float, r);
}
// max across the 16 lanes of a row via row_ror:1,2,4,8
static __device__ __forceinline__ float rowmax16(float v) {
    v = fmaxf(v, fdpp<0x121>(v));
    v = fmaxf(v, fdpp<0x122>(v));
    v = fmaxf(v, fdpp<0x124>(v));
    v = fmaxf(v, fdpp<0x128>(v));
    return v;
}

// ---------------------------------------------------------------------------
// Pre-pass: f32 -> bf16 row-major copy (for x)
__global__ void cvt_bf16_kernel(const float* __restrict__ src,
                                bf16_t* __restrict__ dst, int n)
{
    for (int i = blockIdx.x * blockDim.x + threadIdx.x; i < n;
         i += gridDim.x * blockDim.x)
        dst[i] = f2bf(src[i]);
}

// Pre-pass: pack W[K,N] f32 into B-fragment order:
// Wp[(((kc*2 + half)*N) + n)*16 + e] = bf16(W[k][n]),
//   k = kc*32 + 8*half + (e<8 ? e : 16 + (e-8))
__global__ void pack_w_kernel(const float* __restrict__ W,
                              bf16_t* __restrict__ Wp)
{
    const int n_el = D_MODEL * D_MODEL;   // 1M elements
    for (int o = blockIdx.x * blockDim.x + threadIdx.x; o < n_el;
         o += gridDim.x * blockDim.x) {
        const int e    = o & 15;
        const int n    = (o >> 4) & (D_MODEL - 1);
        const int half = (o >> 14) & 1;
        const int kc   = o >> 15;
        const int k    = kc * 32 + 8 * half + (e < 8 ? e : 16 + (e - 8));
        Wp[o] = f2bf(W[(size_t)k * D_MODEL + n]);
    }
}

// ---------------------------------------------------------------------------
// GEMM: C[M,N] = A_bf16[M,K] * Wpack + bias   (bf16 WMMA, f32 accumulate)
// FMODE: 0=no f32 out, 1=flat [M,N], 2=headsplit [B,H,Tout,Dh] at toff+t
// BMODE: 0=no bf16 out, 1=headsplit rowmajor bf16 (scaled), 2=headsplit transposed bf16
template <int FMODE, int BMODE>
__global__ __launch_bounds__(256) void gemm_kernel(
    const bf16_t* __restrict__ A, const bf16_t* __restrict__ Wp,
    const float* __restrict__ bias, float* __restrict__ outF,
    bf16_t* __restrict__ outB, int Tout, int toff, float bscale)
{
    const int lane = threadIdx.x & 31;
    const int wid  = blockIdx.x * (blockDim.x >> 5) + (threadIdx.x >> 5);
    const int tilesN = D_MODEL / 64;            // 16
    const int m0 = (wid / tilesN) * 16;
    const int n0 = (wid % tilesN) * 64;
    if (m0 >= MROWS) return;
    const int half = lane >> 4;
    const int l15  = lane & 15;

    v8f acc[4];
#pragma unroll
    for (int i = 0; i < 4; ++i) acc[i] = vzero8();

    const bf16_t* arow = A + (size_t)(m0 + l15) * D_MODEL;
    for (int kc = 0; kc < D_MODEL / 32; ++kc) {
        const bf16_t* ap = arow + kc * 32 + 8 * half;
        const v16bf a = join8(*(const v8bf*)ap, *(const v8bf*)(ap + 16));
        const bf16_t* wbase =
            Wp + ((size_t)(kc * 2 + half) * D_MODEL + n0 + l15) * 16;
#pragma unroll
        for (int nt = 0; nt < 4; ++nt) {
            const v16bf bfr = *(const v16bf*)(wbase + nt * 16 * 16);
            acc[nt] = __builtin_amdgcn_wmma_f32_16x16x32_bf16(
                false, a, false, bfr, (short)0, acc[nt], false, false);
        }
    }
#pragma unroll
    for (int nt = 0; nt < 4; ++nt) {
        const int n = n0 + nt * 16 + l15;
        const float bv = bias[n];
        const int h = n >> 6, dh = n & 63;
#pragma unroll
        for (int r = 0; r < 8; ++r) {
            const int m = m0 + r + 8 * half;
            const float val = acc[nt][r] + bv;
            const int b = m >> 11, t = m & (TQ - 1);
            if (FMODE == 1)
                outF[(size_t)m * D_MODEL + n] = val;
            else if (FMODE == 2)
                outF[(((size_t)(b * N_HEADS + h) * Tout) + toff + t) * HEAD_DIM + dh] = val;
            if (BMODE == 1)
                outB[(((size_t)(b * N_HEADS + h) * Tout) + toff + t) * HEAD_DIM + dh] =
                    f2bf(val * bscale);
            else if (BMODE == 2)
                outB[(((size_t)(b * N_HEADS + h) * HEAD_DIM) + dh) * Tout + toff + t] =
                    f2bf(val);
        }
    }
}

// ---------------------------------------------------------------------------
// Concat caches: f32 copies into d_out k/v, plus bf16 K rowmajor and bf16 V^T
__global__ void cache_concat_kernel(const float* __restrict__ kc,
                                    const float* __restrict__ vc,
                                    float* __restrict__ ko, float* __restrict__ vo,
                                    bf16_t* __restrict__ kbf, bf16_t* __restrict__ vtbf)
{
    const int n = BATCH * N_HEADS * TCACHE * HEAD_DIM;  // 4M
    for (int i = blockIdx.x * blockDim.x + threadIdx.x; i < n;
         i += gridDim.x * blockDim.x) {
        const int bh   = i >> 17;                 // / (TCACHE*HEAD_DIM)
        const int rest = i & ((1 << 17) - 1);
        const int t    = rest >> 6;
        const int dh   = rest & 63;
        const size_t dst = ((size_t)bh << 18) + rest;   // rows [0,TCACHE) of TK
        const float kv = kc[i], vv = vc[i];
        ko[dst] = kv;
        vo[dst] = vv;
        kbf[dst] = f2bf(kv);
        vtbf[((size_t)bh * HEAD_DIM + dh) * TK + t] = f2bf(vv);
    }
}

// ---------------------------------------------------------------------------
// Flash attention: one wave per 32-row query tile (two 16-row M-tiles sharing
// every K/V fragment); 32 keys per iteration; fully branchless online softmax
// so every accumulator has a single def per iteration (no phi copies).
// 18 WMMAs / iteration: 8 QK^T + 8 P*V + 2 row-sum (ones trick).
__global__ __launch_bounds__(256) void flash_attn_kernel(
    const bf16_t* __restrict__ Q,   // [B,H,TQ,Dh] bf16, pre-scaled by Dh^-0.5
    const bf16_t* __restrict__ K,   // [B,H,TK,Dh] bf16 rowmajor
    const bf16_t* __restrict__ Vt,  // [B,H,Dh,TK] bf16 transposed
    bf16_t* __restrict__ O)         // [B,TQ,D_MODEL] bf16
{
    __shared__ bf16_t pbuf[8][2 * 16 * 32];      // two P tiles per wave
    const int lane = threadIdx.x & 31;
    const int wv   = threadIdx.x >> 5;
    const int gw   = blockIdx.x * 8 + wv;
    const int qt = gw & 63;                      // TQ/32 = 64 tiles
    const int h  = (gw >> 6) & (N_HEADS - 1);
    const int b  = gw >> 10;
    const int q0 = qt * 32;
    const int half = lane >> 4, l15 = lane & 15;
    const float NEG_INF = -__builtin_inff();

    const bf16_t* qp = Q  + ((size_t)(b * N_HEADS + h) * TQ + q0) * HEAD_DIM;
    const bf16_t* kp = K  + (size_t)(b * N_HEADS + h) * TK * HEAD_DIM;
    const bf16_t* vtp= Vt + (size_t)(b * N_HEADS + h) * HEAD_DIM * TK;

    // Q fragments: [mt][kk]
    v16bf qa[2][2];
#pragma unroll
    for (int mt = 0; mt < 2; ++mt)
#pragma unroll
        for (int kk = 0; kk < 2; ++kk) {
            const bf16_t* p =
                qp + (size_t)(mt * 16 + l15) * HEAD_DIM + kk * 32 + 8 * half;
            qa[mt][kk] = join8(*(const v8bf*)p, *(const v8bf*)(p + 16));
        }
    // all-ones B fragment for the row-sum WMMA
    v16bf ones;
#pragma unroll
    for (int e = 0; e < 16; ++e) ones[e] = f2bf(1.0f);

    // o[mt][0..3] = output tiles, o[mt][4] = row-sum accumulator
    v8f o[2][5];
#pragma unroll
    for (int mt = 0; mt < 2; ++mt)
#pragma unroll
        for (int i = 0; i < 5; ++i) o[mt][i] = vzero8();
    float mrow[2][8];
#pragma unroll
    for (int mt = 0; mt < 2; ++mt)
#pragma unroll
        for (int r = 0; r < 8; ++r) mrow[mt][r] = NEG_INF;

    bf16_t* pl = pbuf[wv];
    const int jmax = q0 + 31 + (TK - TQ);        // last valid key (inclusive)
    for (int j0 = 0; j0 <= jmax; j0 += 32) {
        if (j0 + 32 <= jmax) {   // prefetch next key/value tiles (no phi state)
            __builtin_prefetch(kp + (size_t)(j0 + 32 + lane) * HEAD_DIM, 0, 1);
            __builtin_prefetch(vtp + (size_t)lane * TK + j0 + 32, 0, 1);
        }
        // K fragments once, shared by both M-tiles: [jt][kk]
        v16bf bb[2][2];
#pragma unroll
        for (int jt = 0; jt < 2; ++jt) {
            const bf16_t* kr = kp + (size_t)(j0 + jt * 16 + l15) * HEAD_DIM;
#pragma unroll
            for (int kk = 0; kk < 2; ++kk) {
                const bf16_t* p = kr + kk * 32 + 8 * half;
                bb[jt][kk] = join8(*(const v8bf*)p, *(const v8bf*)(p + 16));
            }
        }
#pragma unroll
        for (int mt = 0; mt < 2; ++mt) {
            // S = Q K^T for two 16-key tiles
            v8f s[2];
#pragma unroll
            for (int jt = 0; jt < 2; ++jt) {
                s[jt] = vzero8();
#pragma unroll
                for (int kk = 0; kk < 2; ++kk)
                    s[jt] = __builtin_amdgcn_wmma_f32_16x16x32_bf16(
                        false, qa[mt][kk], false, bb[jt][kk],
                        (short)0, s[jt], false, false);
            }
            // branchless causal mask (v_cmp + v_cndmask, no control flow)
#pragma unroll
            for (int jt = 0; jt < 2; ++jt) {
                const int kj = j0 + jt * 16 + l15;
#pragma unroll
                for (int r = 0; r < 8; ++r) {
                    const int lim = q0 + mt * 16 + r + 8 * half + (TK - TQ);
                    s[jt][r] = (kj > lim) ? NEG_INF : s[jt][r];
                }
            }
            // branchless online softmax: single def per accumulator
            bf16_t* pm = pl + mt * 512;
#pragma unroll
            for (int r = 0; r < 8; ++r) {
                const float v  = rowmax16(fmaxf(s[0][r], s[1][r]));
                const float nm = fmaxf(mrow[mt][r], v);
                const float alpha = __expf(mrow[mt][r] - nm);   // 0 on first tile
                mrow[mt][r] = nm;
#pragma unroll
                for (int nt = 0; nt < 5; ++nt) o[mt][nt][r] *= alpha;
                const int m = r + 8 * half;
                pm[m * 32 + l15]      = f2bf(__expf(s[0][r] - nm));
                pm[m * 32 + 16 + l15] = f2bf(__expf(s[1][r] - nm));
            }
        }
        // re-read both P tiles as WMMA A fragments (same-wave LDS is in order)
        v16bf pa[2];
#pragma unroll
        for (int mt = 0; mt < 2; ++mt) {
            const bf16_t* plr = pl + mt * 512 + l15 * 32 + 8 * half;
            pa[mt] = join8(*(const v8bf*)plr, *(const v8bf*)(plr + 16));
        }
        // O += P (16x32) * V (32x64); V fragments shared by both M-tiles
#pragma unroll
        for (int nt = 0; nt < 4; ++nt) {
            const bf16_t* pc = vtp + (size_t)(nt * 16 + l15) * TK + j0 + 8 * half;
            const v16bf vb = join8(*(const v8bf*)pc, *(const v8bf*)(pc + 16));
#pragma unroll
            for (int mt = 0; mt < 2; ++mt)
                o[mt][nt] = __builtin_amdgcn_wmma_f32_16x16x32_bf16(
                    false, pa[mt], false, vb, (short)0, o[mt][nt], false, false);
        }
#pragma unroll
        for (int mt = 0; mt < 2; ++mt)
            o[mt][4] = __builtin_amdgcn_wmma_f32_16x16x32_bf16(
                false, pa[mt], false, ones, (short)0, o[mt][4], false, false);
    }
    // epilogue: normalize by the row-sum accumulator, write bf16 [B,TQ,H*Dh]
#pragma unroll
    for (int mt = 0; mt < 2; ++mt)
#pragma unroll
        for (int r = 0; r < 8; ++r) {
            const float inv = 1.0f / o[mt][4][r];
            const int m = q0 + mt * 16 + r + 8 * half;
#pragma unroll
            for (int nt = 0; nt < 4; ++nt) {
                O[((size_t)(b * TQ + m)) * D_MODEL + h * HEAD_DIM + nt * 16 + l15] =
                    f2bf(o[mt][nt][r] * inv);
            }
        }
}

// ---------------------------------------------------------------------------
extern "C" void kernel_launch(void* const* d_in, const int* in_sizes, int n_in,
                              void* d_out, int out_size, void* d_ws, size_t ws_size,
                              hipStream_t stream)
{
    (void)in_sizes; (void)n_in; (void)out_size; (void)ws_size;
    const float* x  = (const float*)d_in[0];
    const float* kc = (const float*)d_in[1];
    const float* vc = (const float*)d_in[2];
    const float* Wq = (const float*)d_in[3];
    const float* bq = (const float*)d_in[4];
    const float* Wk = (const float*)d_in[5];
    const float* bk = (const float*)d_in[6];
    const float* Wv = (const float*)d_in[7];
    const float* bv = (const float*)d_in[8];
    const float* Wo = (const float*)d_in[9];
    const float* bo = (const float*)d_in[10];

    float* out_y = (float*)d_out;                                  // [B,TQ,C]
    float* out_k = out_y + (size_t)BATCH * TQ * D_MODEL;           // [B,H,TK,Dh]
    float* out_v = out_k + (size_t)BATCH * N_HEADS * TK * HEAD_DIM;

    // workspace layout (bf16 elements)
    bf16_t* w = (bf16_t*)d_ws;
    bf16_t* x_bf    = w;                       w += (size_t)MROWS * D_MODEL;          // 4M
    bf16_t* wq_pk   = w;                       w += (size_t)D_MODEL * D_MODEL;        // 1M
    bf16_t* wk_pk   = w;                       w += (size_t)D_MODEL * D_MODEL;
    bf16_t* wv_pk   = w;                       w += (size_t)D_MODEL * D_MODEL;
    bf16_t* wo_pk   = w;                       w += (size_t)D_MODEL * D_MODEL;
    bf16_t* q_bf    = w;                       w += (size_t)BATCH * N_HEADS * TQ * HEAD_DIM;
    bf16_t* k_bf    = w;                       w += (size_t)BATCH * N_HEADS * TK * HEAD_DIM;
    bf16_t* vt_bf   = w;                       w += (size_t)BATCH * N_HEADS * TK * HEAD_DIM;
    bf16_t* attn_bf = w;

    const dim3 blk(256);
    const dim3 gemm_grid((MROWS / 16) * (D_MODEL / 64) / 8);       // 512 blocks

    cvt_bf16_kernel<<<2048, blk, 0, stream>>>(x, x_bf, MROWS * D_MODEL);
    pack_w_kernel<<<1024, blk, 0, stream>>>(Wq, wq_pk);
    pack_w_kernel<<<1024, blk, 0, stream>>>(Wk, wk_pk);
    pack_w_kernel<<<1024, blk, 0, stream>>>(Wv, wv_pk);
    pack_w_kernel<<<1024, blk, 0, stream>>>(Wo, wo_pk);

    // Q: bf16 only, scale folded; K: f32 + bf16 rowmajor; V: f32 + bf16 transposed
    gemm_kernel<0, 1><<<gemm_grid, blk, 0, stream>>>(
        x_bf, wq_pk, bq, nullptr, q_bf, TQ, 0, 0.125f);
    gemm_kernel<2, 1><<<gemm_grid, blk, 0, stream>>>(
        x_bf, wk_pk, bk, out_k, k_bf, TK, TCACHE, 1.0f);
    gemm_kernel<2, 2><<<gemm_grid, blk, 0, stream>>>(
        x_bf, wv_pk, bv, out_v, vt_bf, TK, TCACHE, 1.0f);

    cache_concat_kernel<<<2048, blk, 0, stream>>>(kc, vc, out_k, out_v, k_bf, vt_bf);

    flash_attn_kernel<<<(BATCH * N_HEADS * (TQ / 32)) / 8, blk, 0, stream>>>(
        q_bf, k_bf, vt_bf, attn_bf);

    gemm_kernel<1, 0><<<gemm_grid, blk, 0, stream>>>(
        attn_bf, wo_pk, bo, out_y, nullptr, 0, 0, 1.0f);
}
